// VectorQuantizer_80788334837954
// MI455X (gfx1250) — matrix operand order repaired and checked by reading, not compiled, not run on hardware
//
#include <hip/hip_runtime.h>

typedef __attribute__((ext_vector_type(16))) _Float16 v16h;
typedef __attribute__((ext_vector_type(8)))  _Float16 v8h;
typedef __attribute__((ext_vector_type(8)))  float    v8f;
typedef __attribute__((ext_vector_type(4)))  float    f4;
typedef __attribute__((ext_vector_type(4)))  unsigned u4;
typedef int v4i __attribute__((vector_size(16)));
typedef __attribute__((address_space(3))) v4i* lds_v4i;

#define VQ_EMBED   256
#define VQ_CODES   1024
#define VQ_TOKENS  65536           // 16 * 4096
#define VQ_KCHUNKS 8               // 256 / 32
#define VQ_NTILES  64              // 1024 / 16
#define VQ_TILE_H  (16 * VQ_EMBED) // halves per 16-code B tile (8 KB)

#if __has_builtin(__builtin_amdgcn_global_load_async_to_lds_b128)
#define VQ_ASYNC_LDS 1
#else
#define VQ_ASYNC_LDS 0
#endif

__device__ __forceinline__ void vq_wait_async() {
#if VQ_ASYNC_LDS
#if __has_builtin(__builtin_amdgcn_s_wait_asynccnt)
    __builtin_amdgcn_s_wait_asynccnt(0);
#else
    asm volatile("s_wait_asynccnt 0x0" ::: "memory");
#endif
#endif
}

// ---------------------------------------------------------------------------
// Prep: codebook f32 -> f16 (row major) into ws, plus e2[k] = sum(e^2) in f32.
// One wave (32 lanes) per code row; each lane handles 8 channels.
// ---------------------------------------------------------------------------
__global__ void vq_prep_kernel(const float* __restrict__ cb,
                               _Float16* __restrict__ cbh,
                               float* __restrict__ e2) {
    const int wave = (int)((blockIdx.x * blockDim.x + threadIdx.x) >> 5);
    const int lane = (int)(threadIdx.x & 31);
    if (wave >= VQ_CODES) return;

    const float* src = cb + wave * VQ_EMBED + lane * 8;
    f4 p0 = *(const f4*)(src);
    f4 p1 = *(const f4*)(src + 4);

    float s = 0.f;
    v8h h;
#pragma unroll
    for (int i = 0; i < 4; ++i) {
        s += p0[i] * p0[i] + p1[i] * p1[i];
        h[i]     = (_Float16)p0[i];
        h[4 + i] = (_Float16)p1[i];
    }
    *(v8h*)(cbh + wave * VQ_EMBED + lane * 8) = h;

#pragma unroll
    for (int m = 16; m >= 1; m >>= 1)
        s += __shfl_xor(s, m, 32);
    if (lane == 0) e2[wave] = s;
}

// ---------------------------------------------------------------------------
// Main: one wave per 16-token tile; 8 waves/block share B tiles via LDS with
// double-buffered async global->LDS prefetch. WMMA f16->f32 dot products,
// running argmin of (e2 - 2*x.e), then gather + straight-through output.
// ---------------------------------------------------------------------------
__global__ void __launch_bounds__(256)
vq_main_kernel(const float* __restrict__ latents,
               const float* __restrict__ cb,
               const _Float16* __restrict__ cbh,
               const float* __restrict__ e2,
               float* __restrict__ out) {
    __shared__ _Float16 Bbuf[2][VQ_TILE_H];  // 2 x 8 KB double buffer
    __shared__ float    e2s[VQ_CODES];       // 4 KB

    const int tid     = (int)threadIdx.x;
    const int lane    = tid & 31;
    const int wave    = (int)(blockIdx.x * (blockDim.x >> 5) + (tid >> 5));
    const int tokBase = wave << 4;          // 16 tokens per wave
    const int mrow    = lane & 15;
    const int halfSel = lane >> 4;          // 0: lanes 0-15, 1: lanes 16-31

    // ---- e2 into LDS (1024 floats, 4 per thread) --------------------------
    ((f4*)e2s)[tid] = ((const f4*)e2)[tid];

    // ---- B-tile staging: 8 KB = 256 threads x 32 B ------------------------
    auto stage_tile = [&](int nt) {
        const int buf = nt & 1;
        const _Float16* src = cbh + (size_t)nt * VQ_TILE_H + tid * 16;
        _Float16*       dst = &Bbuf[buf][tid * 16];
#if VQ_ASYNC_LDS
        __builtin_amdgcn_global_load_async_to_lds_b128(
            (v4i*)const_cast<_Float16*>(src), (lds_v4i)(v4i*)dst, 0, 0);
        __builtin_amdgcn_global_load_async_to_lds_b128(
            (v4i*)const_cast<_Float16*>(src), (lds_v4i)(v4i*)dst, 16, 0);
#else
        u4 a = ((const u4*)src)[0];
        u4 b = ((const u4*)src)[1];
        ((u4*)dst)[0] = a;
        ((u4*)dst)[1] = b;
#endif
    };

    stage_tile(0);

    // ---- Load + convert A tile (16 x 256) into WMMA A-layout --------------
    // A operand in contiguous k order (same as B): lanes 0-15 hold K = c..c+15 (halves 0-15);
    // lanes 16-31 hold K = c+16..c+31.
    v16h A[VQ_KCHUNKS];
    {
        const float* arow = latents + (size_t)(tokBase + mrow) * VQ_EMBED;
#pragma unroll
        for (int ch = 0; ch < VQ_KCHUNKS; ++ch) {
            const int c0 = ch * 32 + halfSel * 16;
            f4 p0 = *(const f4*)(arow + c0);
            f4 p1 = *(const f4*)(arow + c0 + 4);
            f4 p2 = *(const f4*)(arow + c0 + 8);
            f4 p3 = *(const f4*)(arow + c0 + 12);
            v16h a;
#pragma unroll
            for (int i = 0; i < 4; ++i) {
                a[i]      = (_Float16)p0[i];
                a[4 + i]  = (_Float16)p1[i];
                a[8 + i]  = (_Float16)p2[i];
                a[12 + i] = (_Float16)p3[i];
            }
            A[ch] = a;
        }
    }

    float bestD[8];
    int   bestI[8];
#pragma unroll
    for (int j = 0; j < 8; ++j) { bestD[j] = __builtin_inff(); bestI[j] = 0; }

    // ---- Sweep all 64 tiles of 16 codes -----------------------------------
    for (int nt = 0; nt < VQ_NTILES; ++nt) {
        vq_wait_async();          // our staging portion landed in LDS
        __syncthreads();          // everyone's portion landed; prior reads done
        if (nt + 1 < VQ_NTILES) stage_tile(nt + 1);  // prefetch other buffer

        // B operand in contiguous k order (same as A): lane n = column; lanes 0-15 hold K 0..15,
        // lanes 16-31 hold K 16..31 -> 16 contiguous halves per chunk.
        const _Float16* brow = &Bbuf[nt & 1][0] +
                               (size_t)mrow * VQ_EMBED + halfSel * 16;

        v8f acc = {};
#pragma unroll
        for (int ch = 0; ch < VQ_KCHUNKS; ++ch) {
            v16h b = *(const v16h*)(brow + ch * 32);
            acc = __builtin_amdgcn_wmma_f32_16x16x32_f16(
                false, A[ch], false, b, (short)0, acc, false, false);
        }

        const int   code = nt * 16 + mrow;
        const float e2n  = e2s[code];
#pragma unroll
        for (int j = 0; j < 8; ++j) {
            // ||x-e||^2 up to a per-token constant: e2 - 2*x.e
            float d = fmaf(-2.0f, acc[j], e2n);
            if (d < bestD[j]) { bestD[j] = d; bestI[j] = code; }
        }
    }

    // ---- Cross-lane min within each 16-lane group -------------------------
    // C layout: VGPR j / lanes 0-15 -> token j ; lanes 16-31 -> token j+8.
    // xor masks 1,2,4,8 stay inside each 16-lane group.
#pragma unroll
    for (int j = 0; j < 8; ++j) {
        float d = bestD[j];
        int   i = bestI[j];
#pragma unroll
        for (int m = 1; m < 16; m <<= 1) {
            float od = __shfl_xor(d, m, 32);
            int   oi = __shfl_xor(i, m, 32);
            if (od < d || (od == d && oi < i)) { d = od; i = oi; }
        }
        bestD[j] = d;
        bestI[j] = i;
    }

    // ---- Gather codebook rows, straight-through output --------------------
#pragma unroll
    for (int t = 0; t < 16; ++t) {
        const int idx = __shfl(bestI[t & 7], (t < 8) ? 0 : 16, 32);
        const float* qrow = cb      + (size_t)idx * VQ_EMBED;
        const float* lrow = latents + (size_t)(tokBase + t) * VQ_EMBED;
        float*       orow = out     + (size_t)(tokBase + t) * VQ_EMBED;
#pragma unroll
        for (int it = 0; it < 2; ++it) {
            const int c = it * 128 + lane * 4;
            f4 q = *(const f4*)(qrow + c);
            f4 l = *(const f4*)(lrow + c);
            f4 r = l + (q - l);   // matches reference's l + stopgrad(q - l)
            *(f4*)(orow + c) = r;
        }
    }
}

extern "C" void kernel_launch(void* const* d_in, const int* in_sizes, int n_in,
                              void* d_out, int out_size, void* d_ws, size_t ws_size,
                              hipStream_t stream) {
    (void)in_sizes; (void)n_in; (void)out_size; (void)ws_size;

    const float* latents = (const float*)d_in[0];   // (16,4096,256) f32
    const float* cb      = (const float*)d_in[1];   // (1024,256)    f32
    float*       out     = (float*)d_out;           // (16,4096,256) f32

    // Workspace layout: [0, 512KB) f16 codebook, then 4KB of e2.
    _Float16* cbh = (_Float16*)d_ws;
    float*    e2  = (float*)((char*)d_ws + (size_t)VQ_CODES * VQ_EMBED * sizeof(_Float16));

    // Prep: 1024 waves = 128 blocks * 8 waves.
    vq_prep_kernel<<<128, 256, 0, stream>>>(cb, cbh, e2);

    // Main: 65536/16 = 4096 waves = 512 blocks * 8 waves.
    vq_main_kernel<<<512, 256, 0, stream>>>(latents, cb, cbh, e2, out);
}
